// newLocalGlobalInfoNCE_50362786513096
// MI455X (gfx1250) — compile-verified, matrix operand order checked by reading, not run on hardware
//
#include <hip/hip_runtime.h>

// ---------------- problem constants (from reference setup_inputs) -----------
#define NTOT   8192      // B*P = 8*1024
#define DIM    512
#define NCLS   28
#define KSIM   64
#define TAU    0.07f

// ---------------- WMMA vector types ----------------------------------------
typedef __attribute__((ext_vector_type(16))) __bf16 v16bf;
typedef __attribute__((ext_vector_type(8)))  __bf16 v8bf;
typedef __attribute__((ext_vector_type(8)))  float  v8f;

// Load one 16-element bf16 WMMA fragment chunk pair for this lane.
// Per ISA 16-bit A layout (and mirrored B layout): lane holds row/col = lane%16,
// chunk0 = K[half*8 .. half*8+7], chunk1 = K[half*8+16 .. half*8+23], where the
// caller folds (half*8) into p. Both chunks are 16B-aligned here (D=512).
static __device__ inline v16bf load_frag(const __bf16* p) {
    v8bf lo = *(const v8bf*)(p);
    v8bf hi = *(const v8bf*)(p + 16);
    v16bf r;
#pragma unroll
    for (int i = 0; i < 8; ++i) { r[i] = lo[i]; r[i + 8] = hi[i]; }
    return r;
}

// ---------------- K0: f32 -> bf16 convert + row squared norms ---------------
__launch_bounds__(256)
__global__ void k_prep(const float* __restrict__ S1,
                       __bf16* __restrict__ S1b,
                       float* __restrict__ sq) {
    const int row = blockIdx.x;
    const float* src = S1 + (size_t)row * DIM;
    __bf16* dst = S1b + (size_t)row * DIM;
    float acc = 0.f;
    for (int j = threadIdx.x; j < DIM; j += 256) {
        float v = src[j];
        dst[j] = (__bf16)v;
        acc += v * v;
    }
    __shared__ float red[256];
    red[threadIdx.x] = acc;
    __syncthreads();
    for (int s = 128; s > 0; s >>= 1) {
        if (threadIdx.x < s) red[threadIdx.x] += red[threadIdx.x + s];
        __syncthreads();
    }
    if (threadIdx.x == 0) sq[row] = red[0];
}

// ---------------- K1: fused Gram + distance + per-class sums ----------------
// Block = 256 threads = 8 waves; each wave owns 16 rows (A panel resident in
// 128 VGPRs across the whole column sweep). Processes TWO 16-wide column tiles
// per iteration: two independent WMMA accumulation chains (hides XDL latency,
// consecutive WMMAs are independent) and 2x A-operand reuse per B byte from L2.
__launch_bounds__(256, 1)
__global__ void k_gram(const __bf16* __restrict__ S1b,
                       const float* __restrict__ sq,
                       const int* __restrict__ labels,
                       float* __restrict__ sums /* [NTOT][NCLS] */) {
    __shared__ float lsums[128 * 32];            // 128 rows x 32 (padded) classes
    for (int i = threadIdx.x; i < 128 * 32; i += 256) lsums[i] = 0.f;
    __syncthreads();

    const int wave = threadIdx.x >> 5;
    const int lane = threadIdx.x & 31;
    const int half = lane >> 4;                  // K-offset selector
    const int l16  = lane & 15;                  // row (A) / col (B) within tile
    const int rowBase = blockIdx.x * 128 + wave * 16;

    // Resident A panel: this wave's 16 rows, all of K=512 (16 k-steps).
    v16bf afrag[16];
    {
        const __bf16* arow = S1b + (size_t)(rowBase + l16) * DIM + half * 8;
#pragma unroll
        for (int ks = 0; ks < 16; ++ks) afrag[ks] = load_frag(arow + ks * 32);
    }
    // Squared norms of the 8 result rows this lane produces (C layout: M = half*8+v).
    float sqr[8];
#pragma unroll
    for (int v = 0; v < 8; ++v) sqr[v] = sq[rowBase + half * 8 + v];

    for (int jt = 0; jt < NTOT / 32; ++jt) {
        const int col0 = jt * 32 + l16;
        const int col1 = col0 + 16;
        const __bf16* brow0 = S1b + (size_t)col0 * DIM + half * 8;
        const __bf16* brow1 = S1b + (size_t)col1 * DIM + half * 8;
        v8f acc0 = {};
        v8f acc1 = {};
#pragma unroll
        for (int ks = 0; ks < 16; ++ks) {
            v16bf b0 = load_frag(brow0 + ks * 32);
            v16bf b1 = load_frag(brow1 + ks * 32);
            acc0 = __builtin_amdgcn_wmma_f32_16x16x32_bf16(
                false, afrag[ks], false, b0, (short)0, acc0, false, false);
            acc1 = __builtin_amdgcn_wmma_f32_16x16x32_bf16(
                false, afrag[ks], false, b1, (short)0, acc1, false, false);
        }
        {
            const float sqc = sq[col0];
            const int   cls = labels[col0];
#pragma unroll
            for (int v = 0; v < 8; ++v) {
                float d2   = sqr[v] + sqc - 2.f * acc0[v];
                float dist = sqrtf(fmaxf(d2, 0.f));
                atomicAdd(&lsums[(wave * 16 + half * 8 + v) * 32 + cls], dist);
            }
        }
        {
            const float sqc = sq[col1];
            const int   cls = labels[col1];
#pragma unroll
            for (int v = 0; v < 8; ++v) {
                float d2   = sqr[v] + sqc - 2.f * acc1[v];
                float dist = sqrtf(fmaxf(d2, 0.f));
                atomicAdd(&lsums[(wave * 16 + half * 8 + v) * 32 + cls], dist);
            }
        }
    }
    __syncthreads();
    for (int i = threadIdx.x; i < 128 * NCLS; i += 256) {
        int r = i / NCLS, c = i % NCLS;
        sums[(size_t)(blockIdx.x * 128 + r) * NCLS + c] = lsums[r * 32 + c];
    }
}

// ---------------- K2: per-class argmin (medoid) + centroid gather -----------
__launch_bounds__(256)
__global__ void k_medoid(const float* __restrict__ sums,
                         const int* __restrict__ labels,
                         const __bf16* __restrict__ S1b,
                         __bf16* __restrict__ centB /* [NCLS][DIM] */) {
    const int c = blockIdx.x;                    // 0..27
    float best = 3.402823466e38f;
    int   bi   = 0x7fffffff;
    for (int i = threadIdx.x; i < NTOT; i += 256) {
        if (labels[i] == c) {
            float s = sums[(size_t)i * NCLS + c];
            if (s < best || (s == best && i < bi)) { best = s; bi = i; }
        }
    }
    __shared__ float sv[256];
    __shared__ int   si[256];
    sv[threadIdx.x] = best; si[threadIdx.x] = bi;
    __syncthreads();
    for (int s = 128; s > 0; s >>= 1) {
        if (threadIdx.x < s) {
            float ov = sv[threadIdx.x + s];
            int   oi = si[threadIdx.x + s];
            if (ov < sv[threadIdx.x] || (ov == sv[threadIdx.x] && oi < si[threadIdx.x])) {
                sv[threadIdx.x] = ov; si[threadIdx.x] = oi;
            }
        }
        __syncthreads();
    }
    const int m = si[0];
    for (int j = threadIdx.x; j < DIM; j += 256)
        centB[(size_t)c * DIM + j] = S1b[(size_t)m * DIM + j];
}

// ---------------- K3: logits + weighted CE, per-block partials --------------
// grid = 32 blocks x 256 threads = one thread per row.
__launch_bounds__(256, 1)
__global__ void k_loss(const float* __restrict__ S1,
                       const float* __restrict__ S2,
                       const int* __restrict__ labels,
                       const float* __restrict__ sim,
                       const __bf16* __restrict__ centB,
                       float* __restrict__ partials /* [gridDim][2] */) {
    __shared__ float cent[NCLS * DIM];           // 57.3 KB
    for (int i = threadIdx.x; i < NCLS * DIM; i += 256) cent[i] = (float)centB[i];
    __syncthreads();

    const int row = blockIdx.x * 256 + threadIdx.x;

    float w = 0.f;
    const float* sp = sim + (size_t)row * KSIM;
    for (int k = 0; k < KSIM; ++k) w += sp[k];
    w *= (1.f / (float)KSIM);

    const int lab = labels[row];
    const float* Ss[2] = { S1, S2 };
    float ce[2];
#pragma unroll
    for (int m = 0; m < 2; ++m) {
        const float4* x = (const float4*)(Ss[m] + (size_t)row * DIM);
        float acc[NCLS];
#pragma unroll
        for (int c = 0; c < NCLS; ++c) acc[c] = 0.f;
        for (int j = 0; j < DIM / 4; ++j) {
            float4 a = x[j];
#pragma unroll
            for (int c = 0; c < NCLS; ++c) {
                float4 b = *(const float4*)(cent + c * DIM + j * 4);
                acc[c] += a.x * b.x + a.y * b.y + a.z * b.z + a.w * b.w;
            }
        }
        float mx = acc[0] * (1.f / TAU);
#pragma unroll
        for (int c = 0; c < NCLS; ++c) { acc[c] *= (1.f / TAU); mx = fmaxf(mx, acc[c]); }
        float se = 0.f;
#pragma unroll
        for (int c = 0; c < NCLS; ++c) se += __expf(acc[c] - mx);
        ce[m] = (mx + __logf(se) - acc[lab]) * w;
    }

    __shared__ float r0[256];
    __shared__ float r1[256];
    r0[threadIdx.x] = ce[0];
    r1[threadIdx.x] = ce[1];
    __syncthreads();
    for (int s = 128; s > 0; s >>= 1) {
        if (threadIdx.x < s) {
            r0[threadIdx.x] += r0[threadIdx.x + s];
            r1[threadIdx.x] += r1[threadIdx.x + s];
        }
        __syncthreads();
    }
    if (threadIdx.x == 0) {
        partials[blockIdx.x * 2 + 0] = r0[0];
        partials[blockIdx.x * 2 + 1] = r1[0];
    }
}

// ---------------- K4: final scalar ------------------------------------------
__launch_bounds__(32)
__global__ void k_final(const float* __restrict__ partials, float* __restrict__ out) {
    float l = partials[threadIdx.x * 2 + 0];
    float g = partials[threadIdx.x * 2 + 1];
    for (int s = 16; s > 0; s >>= 1) {
        l += __shfl_down(l, s, 32);
        g += __shfl_down(g, s, 32);
    }
    if (threadIdx.x == 0) {
        float localLoss  = l / (float)NTOT;
        float globalLoss = g / (float)NTOT;
        out[0] = ((1.0f - 0.7f) * localLoss + 0.7f * globalLoss) * 0.5f;
    }
}

// ---------------- launcher ---------------------------------------------------
extern "C" void kernel_launch(void* const* d_in, const int* in_sizes, int n_in,
                              void* d_out, int out_size, void* d_ws, size_t ws_size,
                              hipStream_t stream) {
    const float* S1  = (const float*)d_in[0];   // [N, 512] f32
    const float* S2  = (const float*)d_in[1];   // [N, 512] f32
    const int*   seg = (const int*)  d_in[2];   // [B, P] int32 -> flat [N]
    const float* sim = (const float*)d_in[3];   // [B, P, 64] f32
    float* out = (float*)d_out;

    // workspace carving (hipMalloc base is 256B aligned)
    char* ws = (char*)d_ws;
    __bf16* S1b   = (__bf16*)ws;  ws += (size_t)NTOT * DIM * sizeof(__bf16);  // 8 MB
    float*  sq    = (float*) ws;  ws += (size_t)NTOT * sizeof(float);          // 32 KB
    float*  sums  = (float*) ws;  ws += (size_t)NTOT * NCLS * sizeof(float);   // 896 KB
    float*  parts = (float*) ws;  ws += (size_t)32 * 2 * sizeof(float);
    __bf16* centB = (__bf16*)ws;  ws += (size_t)NCLS * DIM * sizeof(__bf16);   // 28 KB

    k_prep  <<<NTOT,       256, 0, stream>>>(S1, S1b, sq);
    k_gram  <<<NTOT / 128, 256, 0, stream>>>(S1b, sq, seg, sums);
    k_medoid<<<NCLS,       256, 0, stream>>>(sums, seg, S1b, centB);
    k_loss  <<<NTOT / 256, 256, 0, stream>>>(S1, S2, seg, sim, centB, parts);
    k_final <<<1,           32, 0, stream>>>(parts, out);
}
